// CrossRelativeMultiHeadAttention_44212393345712
// MI455X (gfx1250) — matrix operand order repaired and checked by reading, not compiled, not run on hardware
//
#include <hip/hip_runtime.h>

// ---------------------------------------------------------------- types ----
typedef __bf16 bf16_t;
typedef bf16_t v16bf __attribute__((ext_vector_type(16)));
typedef float  v8f   __attribute__((ext_vector_type(8)));
typedef unsigned int v4u __attribute__((ext_vector_type(4)));
typedef int v8i __attribute__((ext_vector_type(8)));
typedef int v4i __attribute__((ext_vector_type(4)));

#define DEV __device__ __forceinline__

#if defined(__has_builtin)
#  if __has_builtin(__builtin_amdgcn_tensor_load_to_lds) && \
      __has_builtin(__builtin_amdgcn_s_wait_tensorcnt) &&   \
      __has_builtin(__builtin_amdgcn_readfirstlane)
#    define HAVE_TDM 1
#  endif
#endif
#ifndef HAVE_TDM
#  define HAVE_TDM 0
#endif

static constexpr int kB  = 2;
static constexpr int kT  = 2048;
static constexpr int kS  = 2048;
static constexpr int kD  = 1024;
static constexpr int kH  = 16;
static constexpr int kDH = 64;
static constexpr int kHD = kH * kDH;    // 1024
static constexpr int kL  = 2 * kT - 1;  // 4095

struct __align__(16) U128 { unsigned int x, y, z, w; };
struct __align__(8)  U64  { unsigned int x, y; };

union FragU { U128 q[2]; v16bf v; };

DEV unsigned short f2bf(float f) {
  unsigned int u = __float_as_uint(f);
  u += 0x7FFFu + ((u >> 16) & 1u);   // round-to-nearest-even
  return (unsigned short)(u >> 16);
}

DEV v16bf load_frag(const unsigned short* p0, const unsigned short* p1) {
  FragU f;
  f.q[0] = *(const U128*)p0;
  f.q[1] = *(const U128*)p1;
  return f.v;
}

DEV v8f wmma_bf16(v16bf a, v16bf b, v8f c) {
  return __builtin_amdgcn_wmma_f32_16x16x32_bf16(false, a, false, b, (short)0, c,
                                                 false, false);
}

DEV v8f v8f_zero() {
  v8f z = {0.f, 0.f, 0.f, 0.f, 0.f, 0.f, 0.f, 0.f};
  return z;
}

#if HAVE_TDM
// Build a 2D tensor D# (D#.data_size = 2 bytes) and issue the DMA.
// Fields per CDNA5 ISA; all dwords forced scalar.
// clang-23 toolchain: 6-arg builtin (g0 v4u, g1 v8i, g2 v4i, g3 v4i, g4 v8i, cpol).
DEV int rfl(unsigned v) { return __builtin_amdgcn_readfirstlane((int)v); }

DEV void tdm_load_2d_bf16(unsigned lds_off, const void* gptr,
                          unsigned tile_d0, unsigned tile_d1,
                          unsigned tensor_d0, unsigned tensor_d1,
                          unsigned stride0) {
  unsigned long long ga = (unsigned long long)(__SIZE_TYPE__)gptr;
  v4u g0;
  g0[0] = (unsigned)rfl(1u);                               // count=1 user D#
  g0[1] = (unsigned)rfl(lds_off);                          // lds_addr
  g0[2] = (unsigned)rfl((unsigned)ga);                     // global_addr lo
  g0[3] = (unsigned)rfl(((unsigned)(ga >> 32) & 0x01FFFFFFu) | (2u << 30));
  v8i g1;
  g1[0] = rfl(1u << 16);                                   // data_size=2B
  g1[1] = rfl(tensor_d0 << 16);                            // tensor_dim0[15:0]
  g1[2] = rfl((tensor_d0 >> 16) | (tensor_d1 << 16));      // dim0 hi | dim1 lo
  g1[3] = rfl((tensor_d1 >> 16) | (tile_d0 << 16));        // dim1 hi | tile0
  g1[4] = rfl(tile_d1 & 0xFFFFu);                          // tile1 | tile2=0
  g1[5] = rfl(stride0);                                    // dim0 stride lo32
  g1[6] = rfl(0u);                                         // stride hi | s1 lo
  g1[7] = rfl(0u);
  v4i g2 = {0, 0, 0, 0};
  v4i g3 = {0, 0, 0, 0};
  v8i g4 = {0, 0, 0, 0, 0, 0, 0, 0};
  __builtin_amdgcn_tensor_load_to_lds(g0, g1, g2, g3, g4, 0);
}
#endif

// ------------------------------------------------------- cast f32 -> bf16 --
__global__ __launch_bounds__(256) void cast_f32_bf16_kernel(
    const float* __restrict__ src, unsigned short* __restrict__ dst, int n4) {
  int i = blockIdx.x * blockDim.x + threadIdx.x;
  if (i >= n4) return;
  float4 v = ((const float4*)src)[i];
  U64 u;
  u.x = (unsigned)f2bf(v.x) | ((unsigned)f2bf(v.y) << 16);
  u.y = (unsigned)f2bf(v.z) | ((unsigned)f2bf(v.w) << 16);
  ((U64*)dst)[i] = u;
}

// ------------------------------------------- layernorm (+f32 out, +bf16) ---
__global__ __launch_bounds__(256) void ln_cast_kernel(
    const float* __restrict__ x, const float* __restrict__ gamma,
    const float* __restrict__ beta, float* __restrict__ xn,
    unsigned short* __restrict__ xnb) {
  __shared__ float s1[256];
  __shared__ float s2[256];
  const int row = blockIdx.x;
  const int tid = threadIdx.x;
  const float4 v = ((const float4*)(x + (size_t)row * kD))[tid];
  s1[tid] = v.x + v.y + v.z + v.w;
  s2[tid] = v.x * v.x + v.y * v.y + v.z * v.z + v.w * v.w;
  __syncthreads();
  for (int o = 128; o > 0; o >>= 1) {
    if (tid < o) { s1[tid] += s1[tid + o]; s2[tid] += s2[tid + o]; }
    __syncthreads();
  }
  const float mu  = s1[0] * (1.0f / kD);
  const float var = s2[0] * (1.0f / kD) - mu * mu;
  const float rs  = rsqrtf(var + 1e-5f);
  const float4 g  = ((const float4*)gamma)[tid];
  const float4 bb = ((const float4*)beta)[tid];
  float4 o;
  o.x = (v.x - mu) * rs * g.x + bb.x;
  o.y = (v.y - mu) * rs * g.y + bb.y;
  o.z = (v.z - mu) * rs * g.z + bb.z;
  o.w = (v.w - mu) * rs * g.w + bb.w;
  ((float4*)(xn + (size_t)row * kD))[tid] = o;
  U64 u;
  u.x = (unsigned)f2bf(o.x) | ((unsigned)f2bf(o.y) << 16);
  u.y = (unsigned)f2bf(o.z) | ((unsigned)f2bf(o.w) << 16);
  ((U64*)(xnb + (size_t)row * kD))[tid] = u;
}

// --------------------------------------------------- tiled bf16 WMMA GEMM --
// C[M,N] = A[M,K] (bf16, row-major) @ Bw[K,N] (bf16, row-major)
// mode 0: Cb = bf16(C)      mode 1: Cf = resid + C + bias (f32)
__global__ __launch_bounds__(128) void gemm_bf16_kernel(
    const unsigned short* __restrict__ A, const unsigned short* __restrict__ Bw,
    unsigned short* __restrict__ Cb, const float* __restrict__ resid,
    const float* __restrict__ bias, float* __restrict__ Cf, int M, int N, int K,
    int mode) {
  __shared__ __align__(16) unsigned short lA[64][40];  // [m][k]
  __shared__ __align__(16) unsigned short lB[64][40];  // [n][k] (transposed)
  const int tid  = threadIdx.x;
  const int lane = tid & 31;
  const int wave = tid >> 5;
  const int m0   = blockIdx.y * 64;
  const int n0   = blockIdx.x * 64;

  v8f acc[4];
#pragma unroll
  for (int n = 0; n < 4; ++n) acc[n] = v8f_zero();

  for (int k0 = 0; k0 < K; k0 += 32) {
    // stage A tile (64x32)
    {
      const int r = tid >> 1, off = (tid & 1) * 16;
      const U128* src = (const U128*)(A + (size_t)(m0 + r) * K + k0 + off);
      *(U128*)&lA[r][off]     = src[0];
      *(U128*)&lA[r][off + 8] = src[1];
    }
    // stage B tile transposed (32x64 -> [n][k])
    {
      const int k = tid >> 2, noff = (tid & 3) * 16;
      U128 d0 = *(const U128*)(Bw + (size_t)(k0 + k) * N + n0 + noff);
      U128 d1 = *(const U128*)(Bw + (size_t)(k0 + k) * N + n0 + noff + 8);
      const unsigned short* p0 = (const unsigned short*)&d0;
      const unsigned short* p1 = (const unsigned short*)&d1;
#pragma unroll
      for (int i = 0; i < 8; ++i) lB[noff + i][k] = p0[i];
#pragma unroll
      for (int i = 0; i < 8; ++i) lB[noff + 8 + i][k] = p1[i];
    }
    __syncthreads();

    const int mrow = wave * 16 + (lane & 15);
    const int ka   = (lane < 16) ? 0 : 16;  // A frag K pattern
    const int kb   = (lane < 16) ? 0 : 16;  // B frag K pattern
    v16bf af = load_frag(&lA[mrow][ka], &lA[mrow][ka + 8]);
    v16bf bf[4];
#pragma unroll
    for (int n = 0; n < 4; ++n) {
      const int ncol = n * 16 + (lane & 15);
      bf[n] = load_frag(&lB[ncol][kb], &lB[ncol][kb + 8]);
    }
#pragma unroll
    for (int n = 0; n < 4; ++n) acc[n] = wmma_bf16(af, bf[n], acc[n]);
    __syncthreads();
  }

  const int half = lane >> 4;
#pragma unroll
  for (int n = 0; n < 4; ++n) {
    const int ncol = n0 + n * 16 + (lane & 15);
#pragma unroll
    for (int r = 0; r < 8; ++r) {
      const int mrow = m0 + wave * 16 + r + 8 * half;
      const float vv = acc[n][r];
      if (mode == 0) {
        Cb[(size_t)mrow * N + ncol] = f2bf(vv);
      } else {
        Cf[(size_t)mrow * N + ncol] =
            resid[(size_t)mrow * N + ncol] + vv + bias[ncol];
      }
    }
  }
}

// ---------------------------------------------- flash attention + rel-pos --
// q/k/v: [B*T, H*DH] bf16.  ew: [2T-1, DH] bf16.  ow: [B*T, H*DH] bf16.
__global__ __launch_bounds__(128) void attn_kernel(
    const unsigned short* __restrict__ qw, const unsigned short* __restrict__ kw,
    const unsigned short* __restrict__ vw, const unsigned short* __restrict__ ew,
    unsigned short* __restrict__ ow) {
  __shared__ __align__(16) unsigned short lK[32][64];    // [s][dh]
  __shared__ __align__(16) unsigned short lV[64][40];    // [dh][s]
  __shared__ __align__(16) unsigned short lE[256][64];   // 4 waves x 64 d rows
  __shared__ __align__(16) unsigned short lP[4][16][40]; // per-wave P (bf16)
  __shared__ float lR[4][16][48];                        // per-wave rel window

  const int tid  = threadIdx.x;
  const int lane = tid & 31;
  const int wave = tid >> 5;
  const int half = lane >> 4;

  const int TT = kT / 64;
  const int t_tile = blockIdx.x % TT;
  const int h = (blockIdx.x / TT) % kH;
  const int b = blockIdx.x / (TT * kH);
  const int t0 = t_tile * 64;
  const int tw = t0 + wave * 16;  // this wave's query base row

  const float scale = 0.125f;  // 1/sqrt(DH)

  // Q fragments straight from global (A-frag layout is contiguous 16B runs).
  const int m  = lane & 15;
  const int ka = (lane < 16) ? 0 : 16;
  const int kb = (lane < 16) ? 0 : 16;
  const unsigned short* qp =
      qw + (size_t)(b * kT + tw + m) * kHD + h * kDH;
  v16bf aq[2];
  aq[0] = load_frag(qp + ka, qp + ka + 8);
  aq[1] = load_frag(qp + 32 + ka, qp + 32 + ka + 8);

  v8f acc_o[4];
#pragma unroll
  for (int n = 0; n < 4; ++n) acc_o[n] = v8f_zero();
  float mrow[8], lrow[8];
#pragma unroll
  for (int r = 0; r < 8; ++r) { mrow[r] = -3.0e38f; lrow[r] = 0.f; }

#if HAVE_TDM
  const unsigned ldsK_off = (unsigned)(__SIZE_TYPE__)&lK[0][0];
  const unsigned ldsE_off = (unsigned)(__SIZE_TYPE__)&lE[0][0];
#endif

  for (int s0 = 0; s0 < kS; s0 += 32) {
    __syncthreads();
#if HAVE_TDM
    // K tile: Tensor Data Mover, 32 rows x 64 bf16, row stride 1024 elems.
    if (wave == 0) {
      tdm_load_2d_bf16(ldsK_off,
                       kw + (size_t)(b * kS + s0) * kHD + h * kDH,
                       /*tile*/ 64, 32, /*tensor*/ 64, (unsigned)kS,
                       /*stride0*/ (unsigned)kHD);
    }
    // Per-wave E window: 64 rows x 64 bf16 contiguous tile.
    {
      int er0 = (kT - 1) + s0 - tw - 15;
      er0 = er0 < 0 ? 0 : (er0 > kL - 64 ? kL - 64 : er0);
      tdm_load_2d_bf16(ldsE_off + (unsigned)wave * 64 * 64 * 2,
                       ew + (size_t)er0 * kDH,
                       /*tile*/ 64, 64, /*tensor*/ 64, (unsigned)kL,
                       /*stride0*/ (unsigned)kDH);
    }
#else
    // stage K tile [s][dh]
    {
      const int s = tid >> 2, off = (tid & 3) * 16;
      const U128* src =
          (const U128*)(kw + (size_t)(b * kS + s0 + s) * kHD + h * kDH + off);
      *(U128*)&lK[s][off]     = src[0];
      *(U128*)&lK[s][off + 8] = src[1];
    }
    // stage per-wave E windows: row = base-15+d, base = T-1+s0-tw(wave)
    {
#pragma unroll
      for (int j = 0; j < 2; ++j) {
        const int row  = tid * 2 + j;
        const int widx = row >> 6;
        const int d    = row & 63;
        int er = (kT - 1) + s0 - (t0 + 16 * widx) - 15 + d;
        er = er < 0 ? 0 : (er > kL - 1 ? kL - 1 : er);
        const U128* src = (const U128*)(ew + (size_t)er * kDH);
        U128* dst = (U128*)&lE[row][0];
#pragma unroll
        for (int q2 = 0; q2 < 8; ++q2) dst[q2] = src[q2];
      }
    }
#endif
    // stage V tile transposed [dh][s] (manual; TDM has no transpose path)
    {
      const int s = tid >> 2, off = (tid & 3) * 16;
      U128 d0 = *(const U128*)(vw + (size_t)(b * kS + s0 + s) * kHD + h * kDH + off);
      U128 d1 = *(const U128*)(vw + (size_t)(b * kS + s0 + s) * kHD + h * kDH + off + 8);
      const unsigned short* p0 = (const unsigned short*)&d0;
      const unsigned short* p1 = (const unsigned short*)&d1;
#pragma unroll
      for (int i = 0; i < 8; ++i) lV[off + i][s] = p0[i];
#pragma unroll
      for (int i = 0; i < 8; ++i) lV[off + 8 + i][s] = p1[i];
    }
#if HAVE_TDM
    __builtin_amdgcn_s_wait_tensorcnt(0);
#endif
    __syncthreads();

    // ---- scores: QK^T (16x32 per wave); batch frags, then WMMA chain ----
    v16bf bk[2][2];
#pragma unroll
    for (int n = 0; n < 2; ++n) {
      const int sc = n * 16 + (lane & 15);
#pragma unroll
      for (int ks = 0; ks < 2; ++ks)
        bk[n][ks] = load_frag(&lK[sc][ks * 32 + kb], &lK[sc][ks * 32 + kb + 8]);
    }
    v8f accs[2];
#pragma unroll
    for (int n = 0; n < 2; ++n) {
      accs[n] = v8f_zero();
#pragma unroll
      for (int ks = 0; ks < 2; ++ks) accs[n] = wmma_bf16(aq[ks], bk[n][ks], accs[n]);
    }
    // ---- rel: Q @ E_win^T (16x48 per wave) ----
    v16bf be[3][2];
#pragma unroll
    for (int n = 0; n < 3; ++n) {
      const int dc = wave * 64 + n * 16 + (lane & 15);
#pragma unroll
      for (int ks = 0; ks < 2; ++ks)
        be[n][ks] = load_frag(&lE[dc][ks * 32 + kb], &lE[dc][ks * 32 + kb + 8]);
    }
    v8f accr[3];
#pragma unroll
    for (int n = 0; n < 3; ++n) {
      accr[n] = v8f_zero();
#pragma unroll
      for (int ks = 0; ks < 2; ++ks) accr[n] = wmma_bf16(aq[ks], be[n][ks], accr[n]);
    }
    // park rel window in LDS for the skew gather
#pragma unroll
    for (int n = 0; n < 3; ++n)
#pragma unroll
      for (int r = 0; r < 8; ++r)
        lR[wave][r + 8 * half][n * 16 + (lane & 15)] = accr[n][r];

    // ---- combine + online softmax ----
    float pv[2][8];
#pragma unroll
    for (int n = 0; n < 2; ++n)
#pragma unroll
      for (int r = 0; r < 8; ++r) {
        const int ti = r + 8 * half;
        const int sj = n * 16 + (lane & 15);
        const float rel = lR[wave][ti][15 + sj - ti];
        pv[n][r] = (accs[n][r] * scale + rel) * scale;
      }
#pragma unroll
    for (int r = 0; r < 8; ++r) {
      float mm = fmaxf(pv[0][r], pv[1][r]);
#pragma unroll
      for (int o = 1; o < 16; o <<= 1) mm = fmaxf(mm, __shfl_xor(mm, o, 32));
      const float mn = fmaxf(mrow[r], mm);
      const float alpha = __expf(mrow[r] - mn);
      mrow[r] = mn;
      lrow[r] *= alpha;
#pragma unroll
      for (int n = 0; n < 4; ++n) acc_o[n][r] *= alpha;
      float ss = 0.f;
#pragma unroll
      for (int n = 0; n < 2; ++n) {
        pv[n][r] = __expf(pv[n][r] - mn);
        ss += pv[n][r];
      }
#pragma unroll
      for (int o = 1; o < 16; o <<= 1) ss += __shfl_xor(ss, o, 32);
      lrow[r] += ss;
    }
    // ---- P -> LDS (bf16) -> A-frag layout ----
#pragma unroll
    for (int n = 0; n < 2; ++n)
#pragma unroll
      for (int r = 0; r < 8; ++r)
        lP[wave][r + 8 * half][n * 16 + (lane & 15)] = f2bf(pv[n][r]);
    v16bf ap = load_frag(&lP[wave][lane & 15][ka], &lP[wave][lane & 15][ka + 8]);
    // ---- O += P @ V ----
    v16bf bv[4];
#pragma unroll
    for (int n = 0; n < 4; ++n) {
      const int dh = n * 16 + (lane & 15);
      bv[n] = load_frag(&lV[dh][kb], &lV[dh][kb + 8]);
    }
#pragma unroll
    for (int n = 0; n < 4; ++n) acc_o[n] = wmma_bf16(ap, bv[n], acc_o[n]);
  }

  // epilogue: normalize and store bf16
#pragma unroll
  for (int r = 0; r < 8; ++r) {
    const float inv = 1.0f / lrow[r];
    const int trow = tw + r + 8 * half;
#pragma unroll
    for (int n = 0; n < 4; ++n) {
      const int dh = n * 16 + (lane & 15);
      ow[(size_t)(b * kT + trow) * kHD + h * kDH + dh] = f2bf(acc_o[n][r] * inv);
    }
  }
}

// ------------------------------------------------------------------ driver -
extern "C" void kernel_launch(void* const* d_in, const int* in_sizes, int n_in,
                              void* d_out, int out_size, void* d_ws,
                              size_t ws_size, hipStream_t stream) {
  (void)in_sizes; (void)n_in; (void)out_size; (void)ws_size;
  const float* x     = (const float*)d_in[0];
  const float* ctx   = (const float*)d_in[1];
  const float* lut   = (const float*)d_in[2];
  const float* Wq    = (const float*)d_in[3];
  const float* Wk    = (const float*)d_in[4];
  const float* Wv    = (const float*)d_in[5];
  const float* Wo    = (const float*)d_in[6];
  const float* bo    = (const float*)d_in[7];
  const float* gamma = (const float*)d_in[8];
  const float* beta  = (const float*)d_in[9];
  float* out = (float*)d_out;

  const int MT = kB * kT;  // 4096
  const int MS = kB * kS;  // 4096

  char* ws = (char*)d_ws;
  size_t off = 0;
  auto alloc = [&](size_t bytes) -> void* {
    void* p = ws + off;
    off += (bytes + 255) & ~(size_t)255;
    return p;
  };
  unsigned short* xnb  = (unsigned short*)alloc((size_t)MT * kD * 2);
  unsigned short* ctxb = (unsigned short*)alloc((size_t)MS * kD * 2);
  unsigned short* wqb  = (unsigned short*)alloc((size_t)kD * kHD * 2);
  unsigned short* wkb  = (unsigned short*)alloc((size_t)kD * kHD * 2);
  unsigned short* wvb  = (unsigned short*)alloc((size_t)kD * kHD * 2);
  unsigned short* wob  = (unsigned short*)alloc((size_t)kHD * kD * 2);
  unsigned short* eb   = (unsigned short*)alloc((size_t)kL * kDH * 2);
  unsigned short* qb   = (unsigned short*)alloc((size_t)MT * kHD * 2);
  unsigned short* kbuf = (unsigned short*)alloc((size_t)MS * kHD * 2);
  unsigned short* vbuf = (unsigned short*)alloc((size_t)MS * kHD * 2);
  unsigned short* obuf = (unsigned short*)alloc((size_t)MT * kHD * 2);

  auto cast = [&](const float* s, unsigned short* d, int n) {
    const int n4 = n / 4;
    cast_f32_bf16_kernel<<<(n4 + 255) / 256, 256, 0, stream>>>(s, d, n4);
  };
  cast(ctx, ctxb, MS * kD);
  cast(Wq, wqb, kD * kHD);
  cast(Wk, wkb, kD * kHD);
  cast(Wv, wvb, kD * kHD);
  cast(Wo, wob, kHD * kD);
  cast(lut, eb, kL * kDH);

  // layernorm: xn -> d_out (residual), bf16 copy -> xnb
  ln_cast_kernel<<<MT, 256, 0, stream>>>(x, gamma, beta, out, xnb);

  // q/k/v projections
  dim3 gg(kHD / 64, MT / 64);
  gemm_bf16_kernel<<<gg, 128, 0, stream>>>(xnb, wqb, qb, nullptr, nullptr,
                                           nullptr, MT, kHD, kD, 0);
  gemm_bf16_kernel<<<gg, 128, 0, stream>>>(ctxb, wkb, kbuf, nullptr, nullptr,
                                           nullptr, MS, kHD, kD, 0);
  gemm_bf16_kernel<<<gg, 128, 0, stream>>>(ctxb, wvb, vbuf, nullptr, nullptr,
                                           nullptr, MS, kHD, kD, 0);

  // fused relative-position flash attention
  attn_kernel<<<kB * kH * (kT / 64), 128, 0, stream>>>(qb, kbuf, vbuf, eb, obuf);

  // output projection + bias + residual (xn already in d_out)
  dim3 go(kD / 64, MT / 64);
  gemm_bf16_kernel<<<go, 128, 0, stream>>>(obuf, wob, nullptr, out, bo, out, MT,
                                           kD, kHD, 1);
}
